// GAT_51067161150196
// MI455X (gfx1250) — matrix operand order repaired and checked by reading, compiled, not run on hardware
//
#include <hip/hip_runtime.h>
#include <hip/hip_bf16.h>
#include <math.h>

typedef __attribute__((ext_vector_type(16))) _Float16 v16h;
typedef __attribute__((ext_vector_type(8)))  float    v8f;

#define NNODES   10000
#define ERAW     160000
#define NHEADS   8
#define DIN      165
#define DH1      128
#define HID      1024      // 8*128
#define KPAD1    192       // DIN padded to 32
#define KT1      6         // 192/32
#define NT1      128       // 2048/16  (Wl1|Wr1 packed side by side)
#define NCOLS1   2048
#define KT2      32        // 1024/32
#define NCOLS2   16        // Wl2|Wr2 packed (8+8)
#define MTILES   625       // 10000/16

// ---------------- layout helpers (WMMA 16-bit operands, contiguous k = 16h + elem, same order for A and B) ----------------
// A-matrix 16x32 f16: lane = m + 16*(klocal>=16), elem = klocal&15
__device__ __forceinline__ void a_map(int klocal, int m, int& lane, int& elem) {
    lane = m + ((klocal >= 16) ? 16 : 0);
    elem = klocal & 15;
}
// B-matrix 32x16 f16: lane = n + 16*(klocal>=16), elem = klocal&15
__device__ __forceinline__ void b_map(int klocal, int n, int& lane, int& elem) {
    lane = n + ((klocal >= 16) ? 16 : 0);
    elem = klocal & 15;
}

// monotone float<->uint map for atomic max on floats
__device__ __forceinline__ unsigned fmap(float f) {
    unsigned u = __float_as_uint(f);
    return (u & 0x80000000u) ? ~u : (u | 0x80000000u);
}
__device__ __forceinline__ float funmap(unsigned u) {
    return (u & 0x80000000u) ? __uint_as_float(u & 0x7fffffffu) : __uint_as_float(~u);
}

__device__ __forceinline__ int esrc(const int* ei, int e) { return (e < ERAW) ? ei[e] : (e - ERAW); }
__device__ __forceinline__ int edst(const int* ei, int e) { return (e < ERAW) ? ei[ERAW + e] : (e - ERAW); }

// ---------------- init ----------------
__global__ void k_init_max(unsigned* m1, unsigned* m2, int n) {
    int i = blockIdx.x * blockDim.x + threadIdx.x;
    if (i < n) { m1[i] = 0x007FFFFFu; m2[i] = 0x007FFFFFu; }  // fmap(-inf)
}

// ---------------- column stats (biased mean/var over rows) ----------------
__global__ void k_col_stats(const float* __restrict__ X, float* mu, float* var, int rows, int cols) {
    __shared__ float ssum[256], ssq[256];
    int c = blockIdx.x;
    float s = 0.f, q = 0.f;
    for (int r = threadIdx.x; r < rows; r += 256) {
        float v = X[(size_t)r * cols + c];
        s += v; q += v * v;
    }
    ssum[threadIdx.x] = s; ssq[threadIdx.x] = q;
    __syncthreads();
    for (int o = 128; o > 0; o >>= 1) {
        if (threadIdx.x < o) { ssum[threadIdx.x] += ssum[threadIdx.x + o]; ssq[threadIdx.x] += ssq[threadIdx.x + o]; }
        __syncthreads();
    }
    if (threadIdx.x == 0) {
        float m = ssum[0] / rows;
        mu[c] = m;
        var[c] = ssq[0] / rows - m * m;
    }
}

// ---------------- batchnorm1 + pack A (f16, WMMA fragment-major, K padded) ----------------
__global__ void k_bn1_pack(const float* __restrict__ x, const float* __restrict__ mu,
                           const float* __restrict__ var, const float* __restrict__ gamma,
                           const float* __restrict__ beta, _Float16* __restrict__ Apk) {
    int idx = blockIdx.x * blockDim.x + threadIdx.x;      // over NNODES*KPAD1
    if (idx >= NNODES * KPAD1) return;
    int r = idx / KPAD1, k = idx % KPAD1;
    float v = 0.f;
    if (k < DIN)
        v = (x[(size_t)r * DIN + k] - mu[k]) * rsqrtf(var[k] + 1e-5f) * gamma[k] + beta[k];
    int mtile = r >> 4, m = r & 15, kt = k >> 5, kl = k & 31;
    int lane, elem; a_map(kl, m, lane, elem);
    Apk[(((size_t)(mtile * KT1 + kt)) * 32 + lane) * 16 + elem] = (_Float16)v;
}

// ---------------- pack B for layer1: [KPAD1 x 2048] from Wl1|Wr1 ----------------
__global__ void k_pack_w1(const float* __restrict__ Wl, const float* __restrict__ Wr,
                          _Float16* __restrict__ Bpk) {
    int idx = blockIdx.x * blockDim.x + threadIdx.x;      // over KT1*NT1*512
    if (idx >= KT1 * NT1 * 512) return;
    int elem = idx & 15, lane = (idx >> 4) & 31;
    int t2 = idx >> 9;
    int nt = t2 & (NT1 - 1), kt = t2 >> 7;
    int kl = ((lane >= 16) ? 16 : 0) + elem;
    int k  = kt * 32 + kl;
    int n  = nt * 16 + (lane & 15);
    float v = 0.f;
    if (k < DIN) v = (n < HID) ? Wl[(size_t)k * HID + n] : Wr[(size_t)k * HID + (n - HID)];
    Bpk[idx] = (_Float16)v;
}

// ---------------- GEMM1: [10000,192]x[192,2048] -> xlr1 (xl|xr), f16 WMMA, f32 acc ----------------
__global__ void k_gemm1(const _Float16* __restrict__ Apk, const _Float16* __restrict__ Bpk,
                        const float* __restrict__ bl, const float* __restrict__ br,
                        float* __restrict__ C) {
    int wave = threadIdx.x >> 5, lane = threadIdx.x & 31;
    int mtile = blockIdx.y;
    int nt0 = (blockIdx.x * 8 + wave) * 4;                // 4 ntiles per wave
    const v8f zero = {0.f,0.f,0.f,0.f,0.f,0.f,0.f,0.f};
    v8f acc0 = zero, acc1 = zero, acc2 = zero, acc3 = zero;
    for (int kt = 0; kt < KT1; ++kt) {
        v16h a = *(const v16h*)(Apk + (((size_t)(mtile * KT1 + kt)) * 32 + lane) * 16);
        const _Float16* bb = Bpk + (((size_t)kt * NT1) * 32) * 16;
        v16h b0 = *(const v16h*)(bb + ((size_t)(nt0 + 0) * 32 + lane) * 16);
        v16h b1 = *(const v16h*)(bb + ((size_t)(nt0 + 1) * 32 + lane) * 16);
        v16h b2 = *(const v16h*)(bb + ((size_t)(nt0 + 2) * 32 + lane) * 16);
        v16h b3 = *(const v16h*)(bb + ((size_t)(nt0 + 3) * 32 + lane) * 16);
        acc0 = __builtin_amdgcn_wmma_f32_16x16x32_f16(false, a, false, b0, (short)0, acc0, false, false);
        acc1 = __builtin_amdgcn_wmma_f32_16x16x32_f16(false, a, false, b1, (short)0, acc1, false, false);
        acc2 = __builtin_amdgcn_wmma_f32_16x16x32_f16(false, a, false, b2, (short)0, acc2, false, false);
        acc3 = __builtin_amdgcn_wmma_f32_16x16x32_f16(false, a, false, b3, (short)0, acc3, false, false);
    }
    int n = lane & 15, mh = (lane >> 4) * 8;
    v8f accs[4] = {acc0, acc1, acc2, acc3};
    for (int j = 0; j < 4; ++j) {
        int col = (nt0 + j) * 16 + n;
        float bias = (col < HID) ? bl[col] : br[col - HID];
        for (int v = 0; v < 8; ++v) {
            int row = mtile * 16 + mh + v;
            C[(size_t)row * NCOLS1 + col] = accs[j][v] + bias;
        }
    }
}

// ---------------- layer1 edge attention logits: 1 block/edge, 1 wave/head ----------------
__global__ void k_edge_logits1(const float* __restrict__ xlr, const int* __restrict__ ei,
                               const float* __restrict__ att, float* __restrict__ logits, int E2) {
    int e = blockIdx.x;
    if (e >= E2) return;
    int s = esrc(ei, e), d = edst(ei, e);
    int t = threadIdx.x;                // 0..255 ; head = t>>5 (wave-aligned)
    int c = t * 4;
    const float4 xl = *(const float4*)(xlr + (size_t)s * NCOLS1 + c);
    const float4 xr = *(const float4*)(xlr + (size_t)d * NCOLS1 + HID + c);
    const float4 aw = *(const float4*)(att + c);   // att[H][128] flat == channel index
    float p = 0.f;
    float v;
    v = xl.x + xr.x; v = v > 0.f ? v : 0.2f * v; p += v * aw.x;
    v = xl.y + xr.y; v = v > 0.f ? v : 0.2f * v; p += v * aw.y;
    v = xl.z + xr.z; v = v > 0.f ? v : 0.2f * v; p += v * aw.z;
    v = xl.w + xr.w; v = v > 0.f ? v : 0.2f * v; p += v * aw.w;
    for (int o = 16; o > 0; o >>= 1) p += __shfl_down(p, o, 32);
    if ((t & 31) == 0) logits[(size_t)e * NHEADS + (t >> 5)] = p;
}

// ---------------- segment max / exp+sum (shared by both layers) ----------------
__global__ void k_seg_max(const float* __restrict__ logits, const int* __restrict__ ei,
                          unsigned* __restrict__ mInt, int E2) {
    int i = blockIdx.x * blockDim.x + threadIdx.x;
    if (i >= E2 * NHEADS) return;
    int e = i >> 3, h = i & 7;
    int d = edst(ei, e);
    atomicMax(&mInt[d * NHEADS + h], fmap(logits[i]));
}

__global__ void k_exp_sum(float* __restrict__ logits, const int* __restrict__ ei,
                          const unsigned* __restrict__ mInt, float* __restrict__ ssum, int E2) {
    int i = blockIdx.x * blockDim.x + threadIdx.x;
    if (i >= E2 * NHEADS) return;
    int e = i >> 3, h = i & 7;
    int d = edst(ei, e);
    float a = __expf(logits[i] - funmap(mInt[d * NHEADS + h]));
    logits[i] = a;
    atomicAdd(&ssum[d * NHEADS + h], a);
}

// ---------------- layer1 weighted aggregation: out1[dst] += xl[src]*alpha ----------------
__global__ void k_aggregate1(const float* __restrict__ xlr, const int* __restrict__ ei,
                             const float* __restrict__ aexp, const float* __restrict__ ssum,
                             float* __restrict__ out1, int E2) {
    int e = blockIdx.x;
    if (e >= E2) return;
    int s = esrc(ei, e), d = edst(ei, e);
    int t = threadIdx.x;
    int c = t * 4, h = t >> 5;
    float alpha = aexp[(size_t)e * NHEADS + h] / (ssum[d * NHEADS + h] + 1e-16f);
    const float4 xl = *(const float4*)(xlr + (size_t)s * NCOLS1 + c);
    float* o = out1 + (size_t)d * HID + c;
    atomicAdd(o + 0, xl.x * alpha);
    atomicAdd(o + 1, xl.y * alpha);
    atomicAdd(o + 2, xl.z * alpha);
    atomicAdd(o + 3, xl.w * alpha);
}

// ---------------- batchnorm2 + leaky(0.01) + pack A2 (bias1 cancels inside BN) ----------------
__global__ void k_bn2_pack(const float* __restrict__ h, const float* __restrict__ mu,
                           const float* __restrict__ var, const float* __restrict__ gamma,
                           const float* __restrict__ beta, _Float16* __restrict__ Apk) {
    int idx = blockIdx.x * blockDim.x + threadIdx.x;    // over NNODES*HID
    if (idx >= NNODES * HID) return;
    int r = idx >> 10, k = idx & (HID - 1);
    float v = (h[idx] - mu[k]) * rsqrtf(var[k] + 1e-5f) * gamma[k] + beta[k];
    v = (v > 0.f) ? v : 0.01f * v;
    int mtile = r >> 4, m = r & 15, kt = k >> 5, kl = k & 31;
    int lane, elem; a_map(kl, m, lane, elem);
    Apk[(((size_t)(mtile * KT2 + kt)) * 32 + lane) * 16 + elem] = (_Float16)v;
}

// ---------------- pack B for layer2: [1024 x 16] from Wl2|Wr2 ----------------
__global__ void k_pack_w2(const float* __restrict__ Wl, const float* __restrict__ Wr,
                          _Float16* __restrict__ Bpk) {
    int idx = blockIdx.x * blockDim.x + threadIdx.x;   // over KT2*512
    if (idx >= KT2 * 512) return;
    int elem = idx & 15, lane = (idx >> 4) & 31, kt = idx >> 9;
    int k = kt * 32 + ((lane >= 16) ? 16 : 0) + elem;
    int n = lane & 15;
    float v = (n < 8) ? Wl[(size_t)k * 8 + n] : Wr[(size_t)k * 8 + (n - 8)];
    Bpk[idx] = (_Float16)v;
}

// ---------------- GEMM2: [10000,1024]x[1024,16] -> xlr2 ----------------
__global__ void k_gemm2(const _Float16* __restrict__ Apk, const _Float16* __restrict__ Bpk,
                        const float* __restrict__ bl, const float* __restrict__ br,
                        float* __restrict__ C) {
    int wave = threadIdx.x >> 5, lane = threadIdx.x & 31;
    int mtile = blockIdx.x * 8 + wave;
    if (mtile >= MTILES) return;                        // whole-wave uniform
    const v8f zero = {0.f,0.f,0.f,0.f,0.f,0.f,0.f,0.f};
    v8f acc = zero;
    for (int kt = 0; kt < KT2; ++kt) {
        v16h a = *(const v16h*)(Apk + (((size_t)(mtile * KT2 + kt)) * 32 + lane) * 16);
        v16h b = *(const v16h*)(Bpk + ((size_t)kt * 32 + lane) * 16);
        acc = __builtin_amdgcn_wmma_f32_16x16x32_f16(false, a, false, b, (short)0, acc, false, false);
    }
    int n = lane & 15, mh = (lane >> 4) * 8;
    float bias = (n < 8) ? bl[n] : br[n - 8];
    for (int v = 0; v < 8; ++v) {
        int row = mtile * 16 + mh + v;
        C[(size_t)row * NCOLS2 + n] = acc[v] + bias;
    }
}

// ---------------- layer2 edge logits: 1 thread/edge, 8 heads ----------------
__global__ void k_edge_logits2(const float* __restrict__ xlr2, const int* __restrict__ ei,
                               const float* __restrict__ att2, float* __restrict__ logits, int E2) {
    int e = blockIdx.x * blockDim.x + threadIdx.x;
    if (e >= E2) return;
    int s = esrc(ei, e), d = edst(ei, e);
    const float* xl = xlr2 + (size_t)s * NCOLS2;
    const float* xr = xlr2 + (size_t)d * NCOLS2 + 8;
#pragma unroll
    for (int h = 0; h < NHEADS; ++h) {
        float v = xl[h] + xr[h];
        v = (v > 0.f) ? v : 0.2f * v;
        logits[(size_t)e * NHEADS + h] = v * att2[h];
    }
}

// ---------------- layer2 aggregation ----------------
__global__ void k_aggregate2(const float* __restrict__ xlr2, const int* __restrict__ ei,
                             const float* __restrict__ aexp, const float* __restrict__ ssum,
                             float* __restrict__ acc2, int E2) {
    int i = blockIdx.x * blockDim.x + threadIdx.x;
    if (i >= E2 * NHEADS) return;
    int e = i >> 3, h = i & 7;
    int s = esrc(ei, e), d = edst(ei, e);
    float alpha = aexp[i] / (ssum[d * NHEADS + h] + 1e-16f);
    atomicAdd(&acc2[d * NHEADS + h], xlr2[(size_t)s * NCOLS2 + h] * alpha);
}

// ---------------- finalize: mean over heads + bias2 ----------------
__global__ void k_finalize(const float* __restrict__ acc2, const float* __restrict__ bias2,
                           float* __restrict__ out) {
    int n = blockIdx.x * blockDim.x + threadIdx.x;
    if (n >= NNODES) return;
    float s = 0.f;
#pragma unroll
    for (int h = 0; h < NHEADS; ++h) s += acc2[n * NHEADS + h];
    out[n] = s * (1.f / NHEADS) + bias2[0];
}

// =====================================================================================
extern "C" void kernel_launch(void* const* d_in, const int* in_sizes, int n_in,
                              void* d_out, int out_size, void* d_ws, size_t ws_size,
                              hipStream_t stream) {
    const float* x      = (const float*)d_in[0];
    const float* gamma1 = (const float*)d_in[1];
    const float* beta1  = (const float*)d_in[2];
    const float* Wl1    = (const float*)d_in[3];
    const float* bl1    = (const float*)d_in[4];
    const float* Wr1    = (const float*)d_in[5];
    const float* br1    = (const float*)d_in[6];
    const float* att1   = (const float*)d_in[7];
    // d_in[8] = bias1 : cancels exactly in batchnorm2, unused
    const float* gamma2 = (const float*)d_in[9];
    const float* beta2  = (const float*)d_in[10];
    const float* Wl2    = (const float*)d_in[11];
    const float* bl2    = (const float*)d_in[12];
    const float* Wr2    = (const float*)d_in[13];
    const float* br2    = (const float*)d_in[14];
    const float* att2   = (const float*)d_in[15];
    const float* bias2  = (const float*)d_in[16];
    const int*   ei     = (const int*)d_in[17];
    float* out = (float*)d_out;

    const int E2 = ERAW + NNODES;

    // ---- workspace carve-out ----
    char* w = (char*)d_ws;
    auto carve = [&](size_t bytes) { void* p = (void*)w; w += (bytes + 255) & ~(size_t)255; return p; };
    float*     mu1   = (float*)carve(DIN * 4);
    float*     var1  = (float*)carve(DIN * 4);
    _Float16*  Apk1  = (_Float16*)carve((size_t)MTILES * KT1 * 512 * 2);
    _Float16*  Bpk1  = (_Float16*)carve((size_t)KT1 * NT1 * 512 * 2);
    float*     xlr1  = (float*)carve((size_t)NNODES * NCOLS1 * 4);
    float*     log1  = (float*)carve((size_t)E2 * NHEADS * 4);
    unsigned*  mI1   = (unsigned*)carve((size_t)NNODES * NHEADS * 4);
    float*     s1    = (float*)carve((size_t)NNODES * NHEADS * 4);
    float*     out1  = (float*)carve((size_t)NNODES * HID * 4);
    float*     mu2   = (float*)carve(HID * 4);
    float*     var2  = (float*)carve(HID * 4);
    _Float16*  Apk2  = (_Float16*)carve((size_t)MTILES * KT2 * 512 * 2);
    _Float16*  Bpk2  = (_Float16*)carve((size_t)KT2 * 512 * 2);
    float*     xlr2  = (float*)carve((size_t)NNODES * NCOLS2 * 4);
    float*     log2  = (float*)carve((size_t)E2 * NHEADS * 4);
    unsigned*  mI2   = (unsigned*)carve((size_t)NNODES * NHEADS * 4);
    float*     s2    = (float*)carve((size_t)NNODES * NHEADS * 4);
    float*     acc2  = (float*)carve((size_t)NNODES * NHEADS * 4);

    // ---- zero / init (graph-capture safe) ----
    hipMemsetAsync(out1, 0, (size_t)NNODES * HID * 4, stream);
    hipMemsetAsync(s1,   0, (size_t)NNODES * NHEADS * 4, stream);
    hipMemsetAsync(s2,   0, (size_t)NNODES * NHEADS * 4, stream);
    hipMemsetAsync(acc2, 0, (size_t)NNODES * NHEADS * 4, stream);
    k_init_max<<<(NNODES * NHEADS + 255) / 256, 256, 0, stream>>>(mI1, mI2, NNODES * NHEADS);

    // ---- layer 1 ----
    k_col_stats<<<DIN, 256, 0, stream>>>(x, mu1, var1, NNODES, DIN);
    k_bn1_pack<<<(NNODES * KPAD1 + 255) / 256, 256, 0, stream>>>(x, mu1, var1, gamma1, beta1, Apk1);
    k_pack_w1<<<(KT1 * NT1 * 512 + 255) / 256, 256, 0, stream>>>(Wl1, Wr1, Bpk1);
    {
        dim3 g(NT1 / 32, MTILES);   // 8 waves/block, 4 ntiles/wave
        k_gemm1<<<g, 256, 0, stream>>>(Apk1, Bpk1, bl1, br1, xlr1);
    }
    k_edge_logits1<<<E2, 256, 0, stream>>>(xlr1, ei, att1, log1, E2);
    k_seg_max<<<(E2 * NHEADS + 255) / 256, 256, 0, stream>>>(log1, ei, mI1, E2);
    k_exp_sum<<<(E2 * NHEADS + 255) / 256, 256, 0, stream>>>(log1, ei, mI1, s1, E2);
    k_aggregate1<<<E2, 256, 0, stream>>>(xlr1, ei, log1, s1, out1, E2);

    // ---- layer 2 ----
    k_col_stats<<<HID, 256, 0, stream>>>(out1, mu2, var2, NNODES, HID);
    k_bn2_pack<<<(NNODES * HID + 255) / 256, 256, 0, stream>>>(out1, mu2, var2, gamma2, beta2, Apk2);
    k_pack_w2<<<(KT2 * 512 + 255) / 256, 256, 0, stream>>>(Wl2, Wr2, Bpk2);
    k_gemm2<<<(MTILES + 7) / 8, 256, 0, stream>>>(Apk2, Bpk2, bl2, br2, xlr2);
    k_edge_logits2<<<(E2 + 255) / 256, 256, 0, stream>>>(xlr2, ei, att2, log2, E2);
    k_seg_max<<<(E2 * NHEADS + 255) / 256, 256, 0, stream>>>(log2, ei, mI2, E2);
    k_exp_sum<<<(E2 * NHEADS + 255) / 256, 256, 0, stream>>>(log2, ei, mI2, s2, E2);
    k_aggregate2<<<(E2 * NHEADS + 255) / 256, 256, 0, stream>>>(xlr2, ei, log2, s2, acc2, E2);
    k_finalize<<<(NNODES + 255) / 256, 256, 0, stream>>>(acc2, bias2, out);
}
